// TNET_LF_59365037965385
// MI455X (gfx1250) — compile-verified, matrix operand order checked by reading, not run on hardware
//
#include <hip/hip_runtime.h>
#include <hip/hip_bf16.h>
#include <math.h>

typedef __attribute__((ext_vector_type(16))) _Float16 v16h;
typedef __attribute__((ext_vector_type(8)))  _Float16 v8h;
typedef __attribute__((ext_vector_type(8)))  float    v8f;
typedef __attribute__((ext_vector_type(4)))  unsigned int u32x4;
typedef __attribute__((ext_vector_type(8)))  int          i32x8;
typedef __attribute__((ext_vector_type(4)))  int          i32x4;

#define BB 128
#define SS 128
#define TT 8
#define EE 300
#define EP 320      // E padded to multiple of 32 for WMMA K
#define HH 512
#define GG 2048     // 4H
#define LL 3
#define C2 1024     // 2H
#define C4 2048     // 4H (concat channels)
#define NC 50       // conv out channels

#if defined(__has_builtin)
#if __has_builtin(__builtin_amdgcn_tensor_load_to_lds) && __has_builtin(__builtin_amdgcn_s_wait_tensorcnt)
#define HAVE_TDM 1
#endif
#endif

// ---------------------------------------------------------------- utilities

__global__ void k_zero(float* p, long n) {
  long i = (long)blockIdx.x * blockDim.x + threadIdx.x;
  if (i < n) p[i] = 0.f;
}

// lengths + reversal index (involution): idx[t] = t < L ? L-1-t : t
__global__ void k_len_idx(const int* tok, int* lens, int* idx, int Sn) {
  __shared__ int cnt;
  int b = blockIdx.x;
  if (threadIdx.x == 0) cnt = 0;
  __syncthreads();
  int t = threadIdx.x;
  if (t < Sn && tok[b * Sn + t] != 0) atomicAdd(&cnt, 1);
  __syncthreads();
  int L = cnt;
  if (t < Sn) idx[b * Sn + t] = (t < L) ? (L - 1 - t) : t;
  if (t == 0) lens[b] = L;
}

// embedding gather -> f16, K padded with zeros; optional reversed gather
__global__ void k_embed(const int* tok, const float* emb, const int* idx,
                        _Float16* dst, int Sn) {
  int row = blockIdx.x;            // b*Sn + t
  int b = row / Sn, t = row % Sn;
  int src_t = idx ? idx[b * Sn + t] : t;
  int tk = tok[b * Sn + src_t];
  const float* e = emb + (long)tk * EE;
  _Float16* d = dst + (long)row * EP;
  for (int j = threadIdx.x; j < EP; j += blockDim.x)
    d[j] = (j < EE) ? (_Float16)e[j] : (_Float16)0.f;
}

// f32 -> f16 with column padding (sc source cols -> dc dest cols)
__global__ void k_f2h(const float* src, _Float16* dst, int rows, int sc, int dc) {
  long i = (long)blockIdx.x * blockDim.x + threadIdx.x;
  long n = (long)rows * dc;
  if (i >= n) return;
  int r = (int)(i / dc), c = (int)(i % dc);
  dst[i] = (c < sc) ? (_Float16)src[(long)r * sc + c] : (_Float16)0.f;
}

// -------------------------------------------------- TDM warm (CDNA5 tensor DMA)
// Warms L2/LDS with a 16 x 512 f16 tile of the recurrent weights via the
// Tensor Data Mover before the sequential scan starts. D# per ISA 8.3-8.5.
__global__ void k_tdm_warm(const _Float16* __restrict__ src, float* __restrict__ sink) {
#if HAVE_TDM
  __shared__ _Float16 stage[16 * HH];          // 16 KB tile
  unsigned long ga = (unsigned long)src;
  // D# group 0: count=1, lds_addr=0, global_addr, type=2 ("image")
  u32x4 g0 = { 1u,
               0u,
               (unsigned int)(ga & 0xffffffffu),
               (unsigned int)((ga >> 32) & 0x01ffffffu) | (2u << 30) };
  // D# group 1: wg_mask=0, data_size=1 (2B), tensor 512 x 16, tile 512 x 16,
  // dim0 stride = 512 elements.
  i32x8 g1 = { (int)(1u << 16),        // data_size=2B
               (int)(512u << 16),      // tensor_dim0[15:0] in [63:48]
               (int)(16u << 16),       // tensor_dim1[15:0] in [95:80]
               (int)(512u << 16),      // tile_dim0 in [127:112]
               16,                     // tile_dim1 in [143:128]
               512,                    // tensor_dim0_stride[31:0]
               0, 0 };
  i32x4 gz = { 0, 0, 0, 0 };
#if __clang_major__ >= 23
  i32x8 gz8 = { 0, 0, 0, 0, 0, 0, 0, 0 };
  __builtin_amdgcn_tensor_load_to_lds(g0, g1, gz, gz, gz8, 0);
#else
  __builtin_amdgcn_tensor_load_to_lds(g0, g1, gz, gz, 0);
#endif
  __builtin_amdgcn_s_wait_tensorcnt(0);
  sink[threadIdx.x] = (float)stage[threadIdx.x];
#else
  (void)src; (void)sink;
#endif
}

// ------------------------------------------------------- WMMA GEMM (f16->f32)
// C[M,N] (+)= A[M,K] * B^T, B stored N x K row-major (contiguous K).
// blockDim = 128 (4 waves). Each wave computes a 16x32 output slab
// (two 16x16 tiles, n0 and n0+64, sharing one A fragment -> 2 wmma / K-step).
// grid = (N/128, M/16, batch). sA/sB/sC are per-batch element strides.
__global__ void k_wmma_gemm(const _Float16* __restrict__ A,
                            const _Float16* __restrict__ B,
                            float* __restrict__ C,
                            int K, int lda, int ldb, int ldc,
                            long sA, long sB, long sC, int accum) {
  int lane = threadIdx.x & 31;
  int wave = threadIdx.x >> 5;
  int n0a = blockIdx.x * 128 + wave * 16;
  int n0b = n0a + 64;
  int m0  = blockIdx.y * 16;
  const _Float16* Ab = A + (long)blockIdx.z * sA;
  const _Float16* Bb = B + (long)blockIdx.z * sB;
  float*          Cb = C + (long)blockIdx.z * sC;

  int r  = lane & 15;          // tile row (A: M-row, B: N-col)
  int kh = (lane >> 4) * 8;    // K sub-segment per CDNA5 16-bit layout

  const _Float16* pa  = Ab + (m0 + r) * lda + kh;
  const _Float16* pb0 = Bb + (n0a + r) * ldb + kh;
  const _Float16* pb1 = Bb + (n0b + r) * ldb + kh;

  v8f acc0 = {};
  v8f acc1 = {};
  for (int k0 = 0; k0 < K; k0 += 32) {
    v8h alo = *(const v8h*)pa;
    v8h ahi = *(const v8h*)(pa + 16);
    v8h b0l = *(const v8h*)pb0;
    v8h b0h = *(const v8h*)(pb0 + 16);
    v8h b1l = *(const v8h*)pb1;
    v8h b1h = *(const v8h*)(pb1 + 16);
    v16h a, b0, b1;
#pragma unroll
    for (int i = 0; i < 8; ++i) {
      a[i] = alo[i];  a[8 + i] = ahi[i];
      b0[i] = b0l[i]; b0[8 + i] = b0h[i];
      b1[i] = b1l[i]; b1[8 + i] = b1h[i];
    }
    __builtin_prefetch(pa + 128, 0, 1);   // global_prefetch_b8 next K chunks
    acc0 = __builtin_amdgcn_wmma_f32_16x16x32_f16(false, a, false, b0,
                                                  (short)0, acc0, false, false);
    acc1 = __builtin_amdgcn_wmma_f32_16x16x32_f16(false, a, false, b1,
                                                  (short)0, acc1, false, false);
    pa += 32; pb0 += 32; pb1 += 32;
  }

  // C layout: VGPR q -> M = m0+mo+q, N = n0 + (lane&15)
  int mo = (lane >> 4) * 8;
  float* cp0 = Cb + (m0 + mo) * ldc + n0a + r;
  float* cp1 = Cb + (m0 + mo) * ldc + n0b + r;
  if (accum) {
#pragma unroll
    for (int q = 0; q < 8; ++q) { cp0[q * ldc] += acc0[q]; cp1[q * ldc] += acc1[q]; }
  } else {
#pragma unroll
    for (int q = 0; q < 8; ++q) { cp0[q * ldc] = acc0[q]; cp1[q * ldc] = acc1[q]; }
  }
}

// --------------------------------------------------------------- LSTM gates

__device__ __forceinline__ float sigm(float x) { return 1.f / (1.f + __expf(-x)); }

// gates[B,G] = x_t@Wih^T + h@Whh^T (bias added here). Updates c, writes h
// (f32 into out[b, coff+j, pos] and f16 for the next recurrent GEMM).
__global__ void k_lstm_gate(const float* __restrict__ gates,
                            const float* __restrict__ bias,
                            float* __restrict__ c32,
                            _Float16* __restrict__ h16,
                            float* __restrict__ out,
                            int coff, int Sout, int t, const int* posIdx) {
  int j = blockIdx.y * blockDim.x + threadIdx.x;   // hidden unit
  int b = blockIdx.x;
  if (j >= HH) return;
  const float* g = gates + b * GG;
  float gi = g[j]          + bias[j];
  float gf = g[HH + j]     + bias[HH + j];
  float gg = g[2 * HH + j] + bias[2 * HH + j];
  float go = g[3 * HH + j] + bias[3 * HH + j];
  float c = sigm(gf) * c32[b * HH + j] + sigm(gi) * tanhf(gg);
  float h = sigm(go) * tanhf(c);
  c32[b * HH + j] = c;
  h16[b * HH + j] = (_Float16)h;
  int pos = posIdx ? posIdx[b * Sout + t] : t;
  out[(long)b * (2 * HH) * Sout + (coff + j) * Sout + pos] = h;
}

// zero out positions past each sequence length: v is [B, C2, Sn]
__global__ void k_mask(float* v, const int* lens, int Sn) {
  long i = (long)blockIdx.x * blockDim.x + threadIdx.x;
  long n = (long)BB * C2 * Sn;
  if (i >= n) return;
  int s = (int)(i % Sn);
  int b = (int)(i / ((long)C2 * Sn));
  if (s >= lens[b]) v[i] = 0.f;
}

// ------------------------------------------------------------ CPT machinery

__global__ void k_wpos(const int* span, float* w) {
  int b = blockIdx.x, s = threadIdx.x;
  int p0 = span[b * 2], p1 = span[b * 2 + 1];
  float val = (s < p1) ? (1.f - (float)(p1 - s) / 40.f)
                       : (1.f - (float)(s - p0) / 40.f);
  w[b * SS + s] = val;
}

// scores[b,t,s] = sum_c e[b,c,t] v[b,c,s]; softmax over t (size 8) -> a
__global__ void k_attn(const float* __restrict__ e, const float* __restrict__ v,
                       float* __restrict__ a) {
  int b = blockIdx.x, s = threadIdx.x;   // blockDim = SS
  float sc[TT];
  const float* eb = e + (long)b * C2 * TT;
  const float* vb = v + (long)b * C2 * SS + s;
#pragma unroll
  for (int t = 0; t < TT; ++t) sc[t] = 0.f;
  for (int c = 0; c < C2; ++c) {
    float vv = vb[c * SS];
    const float* ep = eb + c * TT;
#pragma unroll
    for (int t = 0; t < TT; ++t) sc[t] += ep[t] * vv;
  }
  float m = sc[0];
#pragma unroll
  for (int t = 1; t < TT; ++t) m = fmaxf(m, sc[t]);
  float sum = 0.f;
#pragma unroll
  for (int t = 0; t < TT; ++t) { sc[t] = __expf(sc[t] - m); sum += sc[t]; }
  float inv = 1.f / sum;
#pragma unroll
  for (int t = 0; t < TT; ++t) a[(long)b * TT * SS + t * SS + s] = sc[t] * inv;
}

// mid[b,c,s] = sum_t e[b,c,t] a[b,t,s]; fused concat, written TRANSPOSED as
// catT16[b][s][c'] (c'=0..2047: [mid ; v]) so the fc1 GEMM's B operand is
// N x K row-major with contiguous K (b128 fragment loads, no strided path).
__global__ void k_mid_cat(const float* __restrict__ e, const float* __restrict__ a,
                          const float* __restrict__ v, _Float16* __restrict__ catT16) {
  long i = (long)blockIdx.x * blockDim.x + threadIdx.x;
  long n = (long)BB * C2 * SS;
  if (i >= n) return;
  int s = (int)(i % SS);
  int c = (int)((i / SS) % C2);
  int b = (int)(i / ((long)C2 * SS));
  const float* ep = e + (long)b * C2 * TT + c * TT;
  const float* ap = a + (long)b * TT * SS + s;
  float acc = 0.f;
#pragma unroll
  for (int t = 0; t < TT; ++t) acc += ep[t] * ap[t * SS];
  _Float16* cb = catT16 + (long)b * SS * C4 + s * C4;
  cb[c]      = (_Float16)acc;
  cb[C2 + c] = (_Float16)v[i];
}

// v = (relu(mid2 + fc1_b) + v) * w
__global__ void k_cpt(const float* __restrict__ mid2, const float* __restrict__ b1,
                      const float* __restrict__ w, float* __restrict__ v) {
  long i = (long)blockIdx.x * blockDim.x + threadIdx.x;
  long n = (long)BB * C2 * SS;
  if (i >= n) return;
  int s = (int)(i % SS);
  int c = (int)((i / SS) % C2);
  int b = (int)(i / ((long)C2 * SS));
  float x = fmaxf(mid2[i] + b1[c], 0.f);
  v[i] = (x + v[i]) * w[b * SS + s];
}

// conv1d (k=3, pad 1) + bias + relu + max over s. grid = B*NC, block = SS
__global__ void k_conv_max(const float* __restrict__ v, const float* __restrict__ cw,
                           const float* __restrict__ cb, float* __restrict__ z) {
  int bo = blockIdx.x;
  int b = bo / NC, oc = bo % NC;
  int s = threadIdx.x;
  const float* vb = v + (long)b * C2 * SS;
  const float* wb = cw + oc * C2 * 3;
  float acc = 0.f;
  for (int ic = 0; ic < C2; ++ic) {
    const float* vr = vb + ic * SS;
    const float* wr = wb + ic * 3;
#pragma unroll
    for (int k = 0; k < 3; ++k) {
      int ss = s + k - 1;
      if (ss >= 0 && ss < SS) acc += vr[ss] * wr[k];
    }
  }
  acc = fmaxf(acc + cb[oc], 0.f);
  __shared__ float red[SS];
  red[s] = acc;
  __syncthreads();
  for (int st = SS / 2; st > 0; st >>= 1) {
    if (s < st) red[s] = fmaxf(red[s], red[s + st]);
    __syncthreads();
  }
  if (s == 0) z[bo] = red[0];
}

// logits = z @ fc_w^T + fc_b ; log_softmax ; mean NLL. out = [loss, logits...]
__global__ void k_loss(const float* __restrict__ z, const float* __restrict__ fw,
                       const float* __restrict__ fb, const int* __restrict__ label,
                       float* __restrict__ out) {
  int b = threadIdx.x;   // 128 threads, 1 block
  float l[LL];
#pragma unroll
  for (int j = 0; j < LL; ++j) {
    float acc = fb[j];
    for (int k = 0; k < NC; ++k) acc += z[b * NC + k] * fw[j * NC + k];
    l[j] = acc;
  }
  float m = fmaxf(fmaxf(l[0], l[1]), l[2]);
  float sum = 0.f;
#pragma unroll
  for (int j = 0; j < LL; ++j) sum += __expf(l[j] - m);
  float lse = m + __logf(sum);
#pragma unroll
  for (int j = 0; j < LL; ++j) out[1 + b * LL + j] = l[j];
  float lp = l[label[b]] - lse;
  atomicAdd(out, -lp / (float)BB);
}

// ------------------------------------------------------------------- driver

extern "C" void kernel_launch(void* const* d_in, const int* in_sizes, int n_in,
                              void* d_out, int out_size, void* d_ws, size_t ws_size,
                              hipStream_t stream) {
  (void)in_sizes; (void)n_in; (void)out_size; (void)ws_size;

  const int*   raw_text = (const int*)d_in[0];
  const int*   target   = (const int*)d_in[1];
  const int*   span     = (const int*)d_in[2];
  const int*   label    = (const int*)d_in[3];
  const float* emb_W    = (const float*)d_in[4];
  const float* Wih_f32[4] = { (const float*)d_in[5], (const float*)d_in[8],
                              (const float*)d_in[11], (const float*)d_in[14] };
  const float* Whh_f32[4] = { (const float*)d_in[6], (const float*)d_in[9],
                              (const float*)d_in[12], (const float*)d_in[15] };
  const float* bias_v[4]  = { (const float*)d_in[7], (const float*)d_in[10],
                              (const float*)d_in[13], (const float*)d_in[16] };
  const float* conv_w = (const float*)d_in[17];
  const float* conv_b = (const float*)d_in[18];
  const float* fc1_w  = (const float*)d_in[19];
  const float* fc1_b  = (const float*)d_in[20];
  const float* fc_w   = (const float*)d_in[21];
  const float* fc_b   = (const float*)d_in[22];
  float* out = (float*)d_out;

  // bump allocator on workspace
  char* wsp = (char*)d_ws;
  auto alloc = [&](size_t bytes) -> void* {
    void* p = (void*)wsp;
    wsp += (bytes + 255) & ~(size_t)255;
    return p;
  };

  int* lensF = (int*)alloc(BB * 4);
  int* lensA = (int*)alloc(BB * 4);
  int* idxS  = (int*)alloc((size_t)BB * SS * 4);
  int* idxT  = (int*)alloc((size_t)BB * TT * 4);

  _Float16* feat16  = (_Float16*)alloc((size_t)BB * SS * EP * 2);
  _Float16* feat16r = (_Float16*)alloc((size_t)BB * SS * EP * 2);
  _Float16* asp16   = (_Float16*)alloc((size_t)BB * TT * EP * 2);
  _Float16* asp16r  = (_Float16*)alloc((size_t)BB * TT * EP * 2);

  _Float16* wih16[4];
  _Float16* whh16[4];
  for (int i = 0; i < 4; ++i) {
    wih16[i] = (_Float16*)alloc((size_t)GG * EP * 2);
    whh16[i] = (_Float16*)alloc((size_t)GG * HH * 2);
  }
  _Float16* fc1w16 = (_Float16*)alloc((size_t)C2 * C4 * 2);

  float*    gates = (float*)alloc((size_t)BB * GG * 4);
  float*    c32   = (float*)alloc((size_t)BB * HH * 4);
  _Float16* h16   = (_Float16*)alloc((size_t)BB * HH * 2);

  float*    v      = (float*)alloc((size_t)BB * C2 * SS * 4);
  float*    e      = (float*)alloc((size_t)BB * C2 * TT * 4);
  float*    attn   = (float*)alloc((size_t)BB * TT * SS * 4);
  float*    wpos   = (float*)alloc((size_t)BB * SS * 4);
  _Float16* catT16 = (_Float16*)alloc((size_t)BB * SS * C4 * 2);
  float*    mid2   = (float*)alloc((size_t)BB * C2 * SS * 4);
  float*    z      = (float*)alloc((size_t)BB * NC * 4);
  float*    tdms   = (float*)alloc(64 * 4);

  // loss accumulator init
  k_zero<<<1, 32, 0, stream>>>(out, 1);

  // lengths + reversal indices
  k_len_idx<<<BB, SS, 0, stream>>>(raw_text, lensF, idxS, SS);
  k_len_idx<<<BB, TT, 0, stream>>>(target,   lensA, idxT, TT);

  // embeddings -> f16 (forward + reversed)
  k_embed<<<BB * SS, 64, 0, stream>>>(raw_text, emb_W, nullptr, feat16,  SS);
  k_embed<<<BB * SS, 64, 0, stream>>>(raw_text, emb_W, idxS,    feat16r, SS);
  k_embed<<<BB * TT, 64, 0, stream>>>(target,   emb_W, nullptr, asp16,   TT);
  k_embed<<<BB * TT, 64, 0, stream>>>(target,   emb_W, idxT,    asp16r,  TT);

  // weights -> f16 (Wih padded E->EP)
  for (int i = 0; i < 4; ++i) {
    long n1 = (long)GG * EP;
    k_f2h<<<(int)((n1 + 255) / 256), 256, 0, stream>>>(Wih_f32[i], wih16[i], GG, EE, EP);
    long n2 = (long)GG * HH;
    k_f2h<<<(int)((n2 + 255) / 256), 256, 0, stream>>>(Whh_f32[i], whh16[i], GG, HH, HH);
  }
  {
    long n3 = (long)C2 * C4;
    k_f2h<<<(int)((n3 + 255) / 256), 256, 0, stream>>>(fc1_w, fc1w16, C2, C4, C4);
  }

  // one LSTM direction (sequential scan; two WMMA GEMMs + gate fuse per step)
  auto run_lstm = [&](const _Float16* x16, int Sn, const _Float16* wih,
                      const _Float16* whh, const float* bias,
                      float* outv, int coff, const int* posIdx) {
    k_zero<<<(BB * HH + 255) / 256, 256, 0, stream>>>(c32, (long)BB * HH);
    k_zero<<<(BB * HH / 2 + 255) / 256, 256, 0, stream>>>((float*)h16, (long)BB * HH / 2);
    k_tdm_warm<<<1, 32, 0, stream>>>(whh, tdms);   // TDM warm of recurrent weights
    dim3 gg1(GG / 128, BB / 16, 1);
    dim3 ggg(BB, HH / 256, 1);
    for (int t = 0; t < Sn; ++t) {
      // gates = x_t @ Wih^T     (A rows strided by Sn*EP, offset t*EP)
      k_wmma_gemm<<<gg1, 128, 0, stream>>>(x16 + (long)t * EP, wih, gates,
                                           EP, Sn * EP, EP, GG, 0, 0, 0, 0);
      // gates += h @ Whh^T
      k_wmma_gemm<<<gg1, 128, 0, stream>>>(h16, whh, gates,
                                           HH, HH, HH, GG, 0, 0, 0, 1);
      k_lstm_gate<<<ggg, 256, 0, stream>>>(gates, bias, c32, h16, outv,
                                           coff, Sn, t, posIdx);
    }
  };

  // BiLSTM layer 1 over feature -> v [B, 2H, S]
  run_lstm(feat16,  SS, wih16[0], whh16[0], bias_v[0], v, 0,  nullptr);
  run_lstm(feat16r, SS, wih16[1], whh16[1], bias_v[1], v, HH, idxS);
  {
    long n = (long)BB * C2 * SS;
    k_mask<<<(int)((n + 255) / 256), 256, 0, stream>>>(v, lensF, SS);
  }

  // BiLSTM layer 2 over aspect -> e [B, 2H, T]
  run_lstm(asp16,  TT, wih16[2], whh16[2], bias_v[2], e, 0,  nullptr);
  run_lstm(asp16r, TT, wih16[3], whh16[3], bias_v[3], e, HH, idxT);
  {
    long n = (long)BB * C2 * TT;
    k_mask<<<(int)((n + 255) / 256), 256, 0, stream>>>(e, lensA, TT);
  }

  // position weights
  k_wpos<<<BB, SS, 0, stream>>>(span, wpos);

  // two CPT iterations
  long nv = (long)BB * C2 * SS;
  for (int it = 0; it < 2; ++it) {
    k_attn<<<BB, SS, 0, stream>>>(e, v, attn);
    k_mid_cat<<<(int)((nv + 255) / 256), 256, 0, stream>>>(e, attn, v, catT16);
    // mid2[b] = fc1_w (1024x2048, NxK) x catT16[b] (128x2048, NxK) ^T-contract
    dim3 gf(SS / 128, C2 / 16, BB);
    k_wmma_gemm<<<gf, 128, 0, stream>>>(fc1w16, catT16, mid2,
                                        C4, C4, C4, SS,
                                        0, (long)SS * C4, (long)C2 * SS, 0);
    k_cpt<<<(int)((nv + 255) / 256), 256, 0, stream>>>(mid2, fc1_b, wpos, v);
  }

  // conv + relu + global max pool
  k_conv_max<<<BB * NC, SS, 0, stream>>>(v, conv_w, conv_b, z);

  // classifier + log-softmax + mean NLL (out[0] = loss, out[1..] = logits)
  k_loss<<<1, BB, 0, stream>>>(z, fc_w, fc_b, label, out);
}
